// CausalSelfAttention_7060926234968
// MI455X (gfx1250) — matrix operand order repaired and checked by reading, not compiled, not run on hardware
//
#include <hip/hip_runtime.h>
#include <hip/hip_bf16.h>

// Problem constants (match the JAX reference)
#define Bn    2
#define Tn    2048
#define Cn    1024
#define Hn    16
#define HDn   64
#define NC3   3072            // 3*C
#define Mn    4096            // B*T

typedef __attribute__((ext_vector_type(16))) __bf16          v16bf;
typedef __attribute__((ext_vector_type(16))) unsigned short  v16us;
typedef __attribute__((ext_vector_type(8)))  unsigned short  v8us;
typedef __attribute__((ext_vector_type(8)))  float           v8f;
typedef __attribute__((ext_vector_type(4)))  unsigned int    v4u;
typedef __attribute__((ext_vector_type(8)))  int             v8i;
typedef __attribute__((ext_vector_type(4)))  int             v4i;

__device__ __forceinline__ unsigned short f2bf(float f) {
  // round-to-nearest-even float32 -> bf16 bits
  unsigned int u = __builtin_bit_cast(unsigned int, f);
  u += 0x7FFFu + ((u >> 16) & 1u);
  return (unsigned short)(u >> 16);
}

__device__ __forceinline__ v8f wmma_bf16(v16us a, v16us b, v8f c) {
  return __builtin_amdgcn_wmma_f32_16x16x32_bf16(
      false, __builtin_bit_cast(v16bf, a),
      false, __builtin_bit_cast(v16bf, b),
      (short)0, c, false, false);
}

__device__ __forceinline__ v16us cat8(v8us lo, v8us hi) {
  return __builtin_shufflevector(lo, hi, 0, 1, 2, 3, 4, 5, 6, 7,
                                         8, 9, 10, 11, 12, 13, 14, 15);
}

#if __has_builtin(__builtin_amdgcn_tensor_load_to_lds)
#define HAVE_TDM 1
#else
#define HAVE_TDM 0
#endif

#if HAVE_TDM
// Issue one 2D TDM tile load: global (row-major, `stride` elements/row, 2B
// elements) -> LDS at byte offset lds_off. LDS rows are padded: after every
// 64 bytes (16 DWORDs, pad_interval=3) insert 16 bytes (4 DWORDs,
// pad_amount=3) => LDS row stride = 80 bytes = [40] ushorts.
__device__ __forceinline__ void tdm_load_2d(const void* gsrc, unsigned lds_off,
                                            unsigned tile_w, unsigned tile_h,
                                            unsigned stride) {
  unsigned long long ga = (unsigned long long)(uintptr_t)gsrc;
  v4u g0;
  g0[0] = 1u;                                   // count=1 (valid user D#)
  g0[1] = lds_off;                              // lds_addr (bytes)
  g0[2] = (unsigned)ga;                         // global_addr[31:0]
  g0[3] = (unsigned)((ga >> 32) & 0x1FFFFFFu)   // global_addr[56:32]
        | (2u << 30);                           // type=2 ("image")
  v8i g1;
  g1[0] = (int)((1u << 16)                      // data_size=1 -> 2 bytes
              | (1u << 20)                      // pad_enable
              | (3u << 22)                      // pad_interval: 16 DWORDs
              | (3u << 25));                    // pad_amount: 4 DWORDs
  g1[1] = (int)((tile_w & 0xFFFFu) << 16);      // tensor_dim0[15:0]
  g1[2] = (int)((tile_w >> 16) | ((tile_h & 0xFFFFu) << 16)); // dim0 hi | dim1 lo
  g1[3] = (int)((tile_h >> 16) | (tile_w << 16));             // dim1 hi | tile_dim0
  g1[4] = (int)(tile_h & 0xFFFFu);              // tile_dim1 | tile_dim2=0
  g1[5] = (int)stride;                          // tensor_dim0_stride[31:0]
  g1[6] = 0;                                    // stride hi | dim1_stride lo
  g1[7] = 0;
  const v4i z4 = {0, 0, 0, 0};
  const v8i z8 = {0, 0, 0, 0, 0, 0, 0, 0};
  __builtin_amdgcn_tensor_load_to_lds(g0, g1, z4, z4, z8, 0);
}
#endif

__device__ __forceinline__ unsigned lds_byte_off(const void* p) {
  return (unsigned)(uintptr_t)p;   // low 32 bits of generic addr = LDS offset
}

// ---------------------------------------------------------------------------
// Pre-pass kernels: bf16 conversion / transposed-conversion of operands.
// ---------------------------------------------------------------------------
__global__ __launch_bounds__(256) void convert_bf16_kernel(
    const float* __restrict__ s, unsigned short* __restrict__ d, int n) {
  const int i = (blockIdx.x * 256 + threadIdx.x) * 4;
  if (i + 3 < n) {
    const float4 f = *(const float4*)(s + i);
    d[i + 0] = f2bf(f.x);
    d[i + 1] = f2bf(f.y);
    d[i + 2] = f2bf(f.z);
    d[i + 3] = f2bf(f.w);
  }
}

// Wt[n][k] = bf16(W[k][n]); W is [K][N] f32 row-major. 32x32 LDS tile.
__global__ __launch_bounds__(256) void transpose_bf16_kernel(
    const float* __restrict__ W, unsigned short* __restrict__ Wt,
    int K, int N) {
  __shared__ unsigned short t[32][33];
  const int n0 = blockIdx.x * 32, k0 = blockIdx.y * 32;
  const int tx = threadIdx.x, ty = threadIdx.y;      // 32 x 8
#pragma unroll
  for (int i = 0; i < 4; ++i) {
    const int kk = ty + i * 8;
    t[kk][tx] = f2bf(W[(size_t)(k0 + kk) * N + n0 + tx]);
  }
  __syncthreads();
#pragma unroll
  for (int i = 0; i < 4; ++i) {
    const int nn = ty + i * 8;
    Wt[(size_t)(n0 + nn) * K + k0 + tx] = t[tx][nn];
  }
}

// ---------------------------------------------------------------------------
// Tiled bf16 GEMM core: D[M][N] tiles from A[M][K] (row-major bf16) and
// Bt[N][K] (N-major bf16). Block: 256 thr = 8 waves; block tile 64M x 128N;
// wave tile 32x32 = 2x2 WMMA tiles; K chunk 32.
// LDS staged via TDM, double-buffered: chunk k+1 streams via the Tensor Data
// Mover while the WMMAs consume chunk k.
// ---------------------------------------------------------------------------
__device__ __forceinline__ void gemm_tile_body(
    const unsigned short* __restrict__ Abf, const unsigned short* __restrict__ Bt,
    int m0, int n0, int Kdim,
    unsigned short (*As)[64][40], unsigned short (*Bs)[128][40],
    int w, int half, int mrow, int ncol, v8f acc[2][2]) {
  const int wmi = (w >> 2) * 32, wni = (w & 3) * 32;
  const int nK = Kdim / 32;
  const int tid = threadIdx.x;

#if HAVE_TDM
  const unsigned ldsA[2] = {lds_byte_off(&As[0][0][0]), lds_byte_off(&As[1][0][0])};
  const unsigned ldsB[2] = {lds_byte_off(&Bs[0][0][0]), lds_byte_off(&Bs[1][0][0])};
  if (w == 0) {  // prologue: stream chunk 0
    tdm_load_2d(Abf + (size_t)m0 * Kdim, ldsA[0], 32u, 64u, (unsigned)Kdim);
    tdm_load_2d(Bt + (size_t)n0 * Kdim, ldsB[0], 32u, 128u, (unsigned)Kdim);
  }
  (void)tid;
#endif

  for (int kt = 0; kt < nK; ++kt) {
    const int cur = kt & 1;
#if HAVE_TDM
    const int nxt = cur ^ 1;
    if (w == 0) {
      if (kt + 1 < nK) {
        // stream chunk kt+1 while chunk kt is consumed below
        const int k1 = (kt + 1) * 32;
        tdm_load_2d(Abf + (size_t)m0 * Kdim + k1, ldsA[nxt], 32u, 64u,
                    (unsigned)Kdim);
        tdm_load_2d(Bt + (size_t)n0 * Kdim + k1, ldsB[nxt], 32u, 128u,
                    (unsigned)Kdim);
        // TDM completes in order: <=2 outstanding => chunk kt has landed
        __builtin_amdgcn_s_wait_tensorcnt(2);
      } else {
        __builtin_amdgcn_s_wait_tensorcnt(0);
      }
    }
#else
    {  // cooperative copy fallback (single logical buffer per parity)
      const int k0 = kt * 32;
      const int seg = (tid & 3) * 8;
      *(v8us*)&As[cur][tid >> 2][seg] =
          *(const v8us*)(Abf + (size_t)(m0 + (tid >> 2)) * Kdim + k0 + seg);
#pragma unroll
      for (int i = 0; i < 2; ++i) {
        const int rr = (tid >> 2) + i * 64;
        *(v8us*)&Bs[cur][rr][seg] =
            *(const v8us*)(Bt + (size_t)(n0 + rr) * Kdim + k0 + seg);
      }
    }
#endif
    __syncthreads();

    v16us fa[2], fb[2];
#pragma unroll
    for (int i = 0; i < 2; ++i) {
      const int row = wmi + i * 16 + mrow;
      const v8us lo = *(const v8us*)&As[cur][row][half * 8];
      const v8us hi = *(const v8us*)&As[cur][row][16 + half * 8];
      fa[i] = cat8(lo, hi);
    }
#pragma unroll
    for (int j = 0; j < 2; ++j) {
      const int col = wni + j * 16 + ncol;
      const v8us lo = *(const v8us*)&Bs[cur][col][half * 8];
      const v8us hi = *(const v8us*)&Bs[cur][col][16 + half * 8];
      fb[j] = cat8(lo, hi);
    }
#pragma unroll
    for (int i = 0; i < 2; ++i)
#pragma unroll
      for (int j = 0; j < 2; ++j)
        acc[i][j] = wmma_bf16(fa[i], fb[j], acc[i][j]);
    // all waves done reading `cur` before wave 0 re-targets it next iter
    __syncthreads();
  }
}

// ---------------------------------------------------------------------------
// Kernel: qkv = x_bf @ W_attn + b_attn -> scatter into Q,K [B,H,T,HD] and
// V transposed [B,H,HD,T] (all bf16).
// ---------------------------------------------------------------------------
__global__ __launch_bounds__(256) void qkv_gemm_kernel(
    const unsigned short* __restrict__ xbf, const unsigned short* __restrict__ Wt,
    const float* __restrict__ bias,
    unsigned short* __restrict__ qp, unsigned short* __restrict__ kp,
    unsigned short* __restrict__ vtp) {
  __shared__ __align__(16) unsigned short As[2][64][40];
  __shared__ __align__(16) unsigned short Bs[2][128][40];
  const int tid = threadIdx.x, lane = tid & 31, w = tid >> 5;
  const int half = lane >> 4, mrow = lane & 15, ncol = lane & 15;
  const int m0 = blockIdx.y * 64, n0 = blockIdx.x * 128;
  const int wmi = (w >> 2) * 32, wni = (w & 3) * 32;

  v8f acc[2][2];
  const v8f vzero = {};
#pragma unroll
  for (int i = 0; i < 2; ++i)
#pragma unroll
    for (int j = 0; j < 2; ++j) acc[i][j] = vzero;

  gemm_tile_body(xbf, Wt, m0, n0, Cn, As, Bs, w, half, mrow, ncol, acc);

#pragma unroll
  for (int i = 0; i < 2; ++i)
#pragma unroll
    for (int j = 0; j < 2; ++j)
#pragma unroll
      for (int r = 0; r < 8; ++r) {
        const int grow = m0 + wmi + i * 16 + r + 8 * half;  // token in [0,B*T)
        const int gcol = n0 + wni + j * 16 + ncol;          // col in [0,3C)
        const float val = acc[i][j][r] + bias[gcol];
        const int sec = gcol >> 10;                         // 0=q,1=k,2=v
        const int cc = gcol & 1023;
        const int hh = cc >> 6, dd = cc & 63;
        const int bb = grow >> 11, tt = grow & 2047;
        const unsigned short bv = f2bf(val);
        const int bh = bb * Hn + hh;
        if (sec == 2) {  // V transposed: [B,H,HD,T]
          vtp[((size_t)bh * HDn + dd) * Tn + tt] = bv;
        } else {
          unsigned short* dst = (sec == 0) ? qp : kp;
          dst[((size_t)bh * Tn + tt) * HDn + dd] = bv;
        }
      }
}

// ---------------------------------------------------------------------------
// Kernel: flash attention. One wave per (b,h, 16-row q tile).
// ---------------------------------------------------------------------------
__global__ __launch_bounds__(256) void attn_kernel(
    const unsigned short* __restrict__ qp, const unsigned short* __restrict__ kp,
    const unsigned short* __restrict__ vtp, unsigned short* __restrict__ yp) {
  __shared__ __align__(16) unsigned short pbuf[8][16][40];
  const int tid = threadIdx.x, lane = tid & 31, w = tid >> 5;
  const int half = lane >> 4, mrow = lane & 15, ncol = lane & 15;
  const int wid = blockIdx.x * 8 + w;
  const int qtile = wid & (Tn / 16 - 1);
  const int bh = wid >> 7;
  const int qbase = qtile * 16;
  const size_t headoff = (size_t)bh * Tn * HDn;
  (void)tid;

  // Q A-fragments (16x64 = two 16x32 bf16 fragments)
  v16us aq[2];
  {
    const unsigned short* qrow = qp + headoff + (size_t)(qbase + mrow) * HDn;
#pragma unroll
    for (int c = 0; c < 2; ++c) {
      const v8us lo = *(const v8us*)(qrow + c * 32 + half * 16);
      const v8us hi = *(const v8us*)(qrow + c * 32 + half * 16 + 8);
      aq[c] = cat8(lo, hi);
    }
  }

  float mi[8], li[8];
  v8f oacc[4];
  const v8f vzero = {};
#pragma unroll
  for (int r = 0; r < 8; ++r) { mi[r] = -1e30f; li[r] = 0.f; }
#pragma unroll
  for (int t = 0; t < 4; ++t) oacc[t] = vzero;

  const float sc = 0.125f;  // 1/sqrt(HD)
  for (int kc = 0; kc <= qbase + 15; kc += 32) {
    // prefetch next chunk's K/V streams (global_prefetch_b8)
    if (kc + 32 <= qbase + 15) {
      __builtin_prefetch(kp + headoff + (size_t)(kc + 32 + ncol) * HDn, 0, 3);
      __builtin_prefetch(vtp + ((size_t)bh * HDn + ncol) * Tn + kc + 32, 0, 3);
    }

    v8f S[2];
#pragma unroll
    for (int j = 0; j < 2; ++j) {
      const int kb = kc + j * 16;
      const unsigned short* krow =
          kp + headoff + (size_t)(kb + ncol) * HDn + half * 16;
      const v16us bk0 = *(const v16us*)krow;         // d = 0..31 slice
      const v16us bk1 = *(const v16us*)(krow + 32);  // d = 32..63 slice
      v8f s = vzero;
      s = wmma_bf16(aq[0], bk0, s);
      s = wmma_bf16(aq[1], bk1, s);
#pragma unroll
      for (int r = 0; r < 8; ++r) {
        const int qg = qbase + r + 8 * half;
        const int kg = kb + ncol;
        s[r] = (kg <= qg) ? s[r] * sc : -1e30f;
      }
      S[j] = s;
    }

    float tm[8];
#pragma unroll
    for (int r = 0; r < 8; ++r) tm[r] = fmaxf(S[0][r], S[1][r]);
#pragma unroll
    for (int off = 8; off >= 1; off >>= 1)
#pragma unroll
      for (int r = 0; r < 8; ++r)
        tm[r] = fmaxf(tm[r], __shfl_xor(tm[r], off, 32));

    float alpha[8], ps[8];
#pragma unroll
    for (int r = 0; r < 8; ++r) {
      const float mn = fmaxf(mi[r], tm[r]);
      alpha[r] = __expf(mi[r] - mn);
      mi[r] = mn;
      const float p0 = __expf(S[0][r] - mn);
      const float p1 = __expf(S[1][r] - mn);
      pbuf[w][r + 8 * half][ncol]      = f2bf(p0);
      pbuf[w][r + 8 * half][16 + ncol] = f2bf(p1);
      ps[r] = p0 + p1;
    }
#pragma unroll
    for (int off = 8; off >= 1; off >>= 1)
#pragma unroll
      for (int r = 0; r < 8; ++r)
        ps[r] += __shfl_xor(ps[r], off, 32);
#pragma unroll
    for (int r = 0; r < 8; ++r) li[r] = li[r] * alpha[r] + ps[r];
#pragma unroll
    for (int t = 0; t < 4; ++t)
#pragma unroll
      for (int r = 0; r < 8; ++r) oacc[t][r] *= alpha[r];

    asm volatile("s_wait_dscnt 0x0" ::: "memory");
    const v8us plo = *(const v8us*)&pbuf[w][mrow][half * 16];
    const v8us phi = *(const v8us*)&pbuf[w][mrow][half * 16 + 8];
    const v16us ap = cat8(plo, phi);

#pragma unroll
    for (int t = 0; t < 4; ++t) {
      // V transposed [B,H,HD,T]: contiguous 16 k-values per lane
      const unsigned short* vrow =
          vtp + ((size_t)bh * HDn + t * 16 + ncol) * Tn + kc + 16 * half;
      const v16us bv = *(const v16us*)vrow;
      oacc[t] = wmma_bf16(ap, bv, oacc[t]);
    }
  }

  const int b = bh >> 4, h = bh & 15;
#pragma unroll
  for (int r = 0; r < 8; ++r) {
    const float inv = 1.0f / li[r];
    const size_t row = (size_t)b * Tn + qbase + r + 8 * half;
#pragma unroll
    for (int t = 0; t < 4; ++t)
      yp[row * Cn + h * HDn + t * 16 + ncol] = f2bf(oacc[t][r] * inv);
  }
}

// ---------------------------------------------------------------------------
// Kernel: out = y_bf @ W_proj + b_proj (fp32 out)
// ---------------------------------------------------------------------------
__global__ __launch_bounds__(256) void proj_gemm_kernel(
    const unsigned short* __restrict__ yp, const unsigned short* __restrict__ Wt,
    const float* __restrict__ bias, float* __restrict__ out) {
  __shared__ __align__(16) unsigned short As[2][64][40];
  __shared__ __align__(16) unsigned short Bs[2][128][40];
  const int tid = threadIdx.x, lane = tid & 31, w = tid >> 5;
  const int half = lane >> 4, mrow = lane & 15, ncol = lane & 15;
  const int m0 = blockIdx.y * 64, n0 = blockIdx.x * 128;
  const int wmi = (w >> 2) * 32, wni = (w & 3) * 32;
  (void)tid;

  v8f acc[2][2];
  const v8f vzero = {};
#pragma unroll
  for (int i = 0; i < 2; ++i)
#pragma unroll
    for (int j = 0; j < 2; ++j) acc[i][j] = vzero;

  gemm_tile_body(yp, Wt, m0, n0, Cn, As, Bs, w, half, mrow, ncol, acc);

#pragma unroll
  for (int i = 0; i < 2; ++i)
#pragma unroll
    for (int j = 0; j < 2; ++j)
#pragma unroll
      for (int r = 0; r < 8; ++r) {
        const int grow = m0 + wmi + i * 16 + r + 8 * half;
        const int gcol = n0 + wni + j * 16 + ncol;
        out[(size_t)grow * Cn + gcol] = acc[i][j][r] + bias[gcol];
      }
}

// ---------------------------------------------------------------------------
extern "C" void kernel_launch(void* const* d_in, const int* in_sizes, int n_in,
                              void* d_out, int out_size, void* d_ws, size_t ws_size,
                              hipStream_t stream) {
  (void)in_sizes; (void)n_in; (void)out_size; (void)ws_size;

  const float* x      = (const float*)d_in[0];   // [B,T,C]
  const float* W_attn = (const float*)d_in[1];   // [C,3C]
  const float* b_attn = (const float*)d_in[2];   // [3C]
  const float* W_proj = (const float*)d_in[3];   // [C,C]
  const float* b_proj = (const float*)d_in[4];   // [C]
  float* out = (float*)d_out;                    // [B,T,C] fp32

  unsigned short* qp  = (unsigned short*)d_ws;             // [B,H,T,HD]
  unsigned short* kp  = qp  + (size_t)Mn * Cn;             // [B,H,T,HD]
  unsigned short* vtp = kp  + (size_t)Mn * Cn;             // [B,H,HD,T]
  unsigned short* yp  = vtp + (size_t)Mn * Cn;             // [B,T,C]
  unsigned short* xbf = yp  + (size_t)Mn * Cn;             // [B*T,C]
  unsigned short* wat = xbf + (size_t)Mn * Cn;             // [3C,C] (transposed)
  unsigned short* wpt = wat + (size_t)NC3 * Cn;            // [C,C]  (transposed)

  convert_bf16_kernel<<<(Mn * Cn / 4 + 255) / 256, 256, 0, stream>>>(
      x, xbf, Mn * Cn);
  {
    dim3 grid(NC3 / 32, Cn / 32), blk(32, 8);
    transpose_bf16_kernel<<<grid, blk, 0, stream>>>(W_attn, wat, Cn, NC3);
  }
  {
    dim3 grid(Cn / 32, Cn / 32), blk(32, 8);
    transpose_bf16_kernel<<<grid, blk, 0, stream>>>(W_proj, wpt, Cn, Cn);
  }
  {
    dim3 grid(NC3 / 128, Mn / 64);
    qkv_gemm_kernel<<<grid, 256, 0, stream>>>(xbf, wat, b_attn, qp, kp, vtp);
  }
  {
    dim3 grid((Bn * Hn * (Tn / 16)) / 8);
    attn_kernel<<<grid, 256, 0, stream>>>(qp, kp, vtp, yp);
  }
  {
    dim3 grid(Cn / 128, Mn / 64);
    proj_gemm_kernel<<<grid, 256, 0, stream>>>(yp, wpt, b_proj, out);
  }
}
